// PLRNN_51625506898351
// MI455X (gfx1250) — compile-verified, matrix operand order checked by reading, not run on hardware
//
#include <hip/hip_runtime.h>

typedef __attribute__((ext_vector_type(16))) _Float16 v16h;
typedef __attribute__((ext_vector_type(8)))  _Float16 v8h;
typedef __attribute__((ext_vector_type(8)))  float    v8f;

namespace {
constexpr int kB    = 128;
constexpr int kT    = 1024;
constexpr int kDX   = 64;
constexpr int kDZ   = 256;
constexpr int kDB   = 32;
constexpr int kRows = 8;                 // batch rows per workgroup (M padded to 16 for WMMA)
constexpr int kWaves   = 16;             // one wave per 16-wide N tile of DZ
constexpr int kThreads = kWaves * 32;    // 512 threads, wave32
constexpr float kClip  = 5.0f;

// be rows padded: 256 -> 264 f16 (528 B) so per-lane fragment gathers rotate LDS banks
constexpr int kBeStride = 264;

// Dynamic LDS layout (bytes)
constexpr int kWOff  = 0;                          // f16 W [DZ][DZ], diag zeroed : 131072
constexpr int kThOff = kWOff + kDZ * kDZ * 2;      // f32 thetaT [DB][DZ]         :  32768
constexpr int kZOff  = kThOff + kDB * kDZ * 4;     // f32 z [kRows][DZ]           :   8192
constexpr int kBeOff = kZOff + kRows * kDZ * 4;    // f16 be [16][kBeStride]      :   8448
constexpr int kSmem  = kBeOff + 16 * kBeStride * 2;
} // namespace

__global__ __launch_bounds__(kThreads)
void plrnn_step_fused(const float* __restrict__ x, const float* __restrict__ z0,
                      const float* __restrict__ AW, const float* __restrict__ h,
                      const float* __restrict__ alphas, const float* __restrict__ thetas,
                      const int* __restrict__ tau_p, float* __restrict__ out)
{
    extern __shared__ char smem[];
    _Float16* w_h  = (_Float16*)(smem + kWOff);
    float*    thT  = (float*)   (smem + kThOff);
    float*    zsh  = (float*)   (smem + kZOff);
    _Float16* be_h = (_Float16*)(smem + kBeOff);

    const int tid  = threadIdx.x;
    const int lane = tid & 31;
    const int wave = tid >> 5;
    const int b0   = blockIdx.x * kRows;
    const int tau  = tau_p[0];

    // ---------------- one-time setup ----------------
    // W = AW with zeroed diagonal, stored row-major [n][k] in f16 (== W^T as the
    // WMMA B operand: lane n-column, K runs along the row).
    for (int idx = tid; idx < kDZ * kDZ; idx += kThreads) {
        const int n = idx >> 8, k = idx & (kDZ - 1);
        const float v = AW[idx];
        w_h[idx] = (_Float16)((n == k) ? 0.0f : v);
    }
    // thetaT[k][i] = thetas[i][k]  -> conflict-free lane-consecutive reads later
    for (int idx = tid; idx < kDB * kDZ; idx += kThreads) {
        const int k = idx >> 8, i = idx & (kDZ - 1);
        thT[idx] = thetas[i * kDB + k];
    }
    // z state rows 0..7
    for (int idx = tid; idx < kRows * kDZ; idx += kThreads) {
        const int m = idx >> 8, i = idx & (kDZ - 1);
        zsh[idx] = z0[(b0 + m) * kDZ + i];
    }
    // be: zero everything once; rows 8..15 stay zero forever (WMMA M padding)
    for (int idx = tid; idx < 16 * kBeStride; idx += kThreads)
        be_h[idx] = (_Float16)0.0f;
    __syncthreads();

    // ---------------- per-wave constants ----------------
    const int n0 = wave * 16;              // this wave's output-column tile
    const int nc = n0 + (lane & 15);       // column carried by this lane (lanes<16 hold D)
    const float a_diag = AW[nc * kDZ + nc];
    const float hval   = h[nc];

    // Pin W fragments in VGPRs for the whole recurrence (8 chunks x v16h = 64 VGPRs).
    // 16-bit B fragment (32x16, K x N): lane<16 -> K = c*32 + {0..7, 16..23},
    // lane>=16 -> K = c*32 + {8..15, 24..31}, column nc.
    v16h bfrag[8];
#pragma unroll
    for (int c = 0; c < 8; ++c) {
        const int koff = c * 32 + ((lane < 16) ? 0 : 8);
        const v8h lo = *(const v8h*)(&w_h[nc * kDZ + koff]);
        const v8h hi = *(const v8h*)(&w_h[nc * kDZ + koff + 16]);
        bfrag[c] = __builtin_shufflevector(lo, hi, 0,1,2,3,4,5,6,7,8,9,10,11,12,13,14,15);
    }

    // basis-expansion mapping: thread -> one column, 4 rows
    const int icol  = tid & (kDZ - 1);
    const int mbase = (tid >> 8) * 4;      // 0 or 4

    int next_force = tau;

    for (int t = 0; t < kT; ++t) {
        // ---- teacher forcing: z[:, :DX] = x[:, t, :] at t = tau, 2tau, ... ----
        if (t == next_force) {
            next_force += tau;
            const int m = tid >> 6;                    // 8 rows x 64 cols = 512 threads
            const int c = tid & (kDX - 1);
            zsh[m * kDZ + c] = x[(size_t)(b0 + m) * kT * kDX + (size_t)t * kDX + c];
            __syncthreads();
        }

        // ---- dendritic basis expansion: be = sum_k alpha_k * relu(z + theta_k) ----
        const float zv0 = zsh[(mbase + 0) * kDZ + icol];
        const float zv1 = zsh[(mbase + 1) * kDZ + icol];
        const float zv2 = zsh[(mbase + 2) * kDZ + icol];
        const float zv3 = zsh[(mbase + 3) * kDZ + icol];
        float a0 = 0.f, a1 = 0.f, a2 = 0.f, a3 = 0.f;
#pragma unroll
        for (int k = 0; k < kDB; ++k) {
            const float th = thT[k * kDZ + icol];   // lane-consecutive: no bank conflicts
            const float al = alphas[k];             // uniform -> SGPR, hoisted
            a0 += al * fmaxf(zv0 + th, 0.0f);
            a1 += al * fmaxf(zv1 + th, 0.0f);
            a2 += al * fmaxf(zv2 + th, 0.0f);
            a3 += al * fmaxf(zv3 + th, 0.0f);
        }
        be_h[(mbase + 0) * kBeStride + icol] = (_Float16)a0;
        be_h[(mbase + 1) * kBeStride + icol] = (_Float16)a1;
        be_h[(mbase + 2) * kBeStride + icol] = (_Float16)a2;
        be_h[(mbase + 3) * kBeStride + icol] = (_Float16)a3;
        __syncthreads();

        // ---- GEMM tile: acc[16x16] = be(16xK) x W^T(Kx16), K = 256 in 8 WMMA steps ----
        v8f acc = {0.f, 0.f, 0.f, 0.f, 0.f, 0.f, 0.f, 0.f};
#pragma unroll
        for (int c = 0; c < 8; ++c) {
            // 16-bit A fragment (16x32): lane<16 -> K = c*32 + {0..7,16..23},
            // lane>=16 -> K = c*32 + {8..15,24..31}, row M = lane&15.
            const int mrow = lane & 15;
            const int koff = c * 32 + ((lane < 16) ? 0 : 8);
            const v8h alo = *(const v8h*)(&be_h[mrow * kBeStride + koff]);
            const v8h ahi = *(const v8h*)(&be_h[mrow * kBeStride + koff + 16]);
            const v16h afrag =
                __builtin_shufflevector(alo, ahi, 0,1,2,3,4,5,6,7,8,9,10,11,12,13,14,15);
            acc = __builtin_amdgcn_wmma_f32_16x16x32_f16(
                      /*neg_a=*/false, afrag, /*neg_b=*/false, bfrag[c],
                      /*c_mod=*/(short)0, acc, /*reuse_a=*/false, /*reuse_b=*/false);
        }

        // ---- z_new = A.*z + be@W^T + h, clip; update state + stream output ----
        // D layout: VGPR r, lanes 0..15 hold (M=r, N=lane). Rows 8..15 are padding.
        if (lane < 16) {
#pragma unroll
            for (int r = 0; r < kRows; ++r) {
                float zn = a_diag * zsh[r * kDZ + nc] + acc[r] + hval;
                zn = fminf(fmaxf(zn, -kClip), kClip);
                zsh[r * kDZ + nc] = zn;
                out[(size_t)(b0 + r) * kT * kDZ + (size_t)t * kDZ + nc] = zn;
            }
        }
        __syncthreads();
    }
}

extern "C" void kernel_launch(void* const* d_in, const int* in_sizes, int n_in,
                              void* d_out, int out_size, void* d_ws, size_t ws_size,
                              hipStream_t stream) {
    (void)in_sizes; (void)n_in; (void)out_size; (void)d_ws; (void)ws_size;
    const float* x      = (const float*)d_in[0];
    const float* z0     = (const float*)d_in[1];
    const float* AW     = (const float*)d_in[2];
    const float* h      = (const float*)d_in[3];
    const float* alphas = (const float*)d_in[4];
    const float* thetas = (const float*)d_in[5];
    const int*   tau    = (const int*)d_in[6];
    float* out = (float*)d_out;

    // ~176 KB dynamic LDS per workgroup (WGP has 320 KB) — raise the per-kernel cap.
    hipFuncSetAttribute((const void*)plrnn_step_fused,
                        hipFuncAttributeMaxDynamicSharedMemorySize, kSmem);

    dim3 grid(kB / kRows);       // 16 workgroups, one per 8-row batch tile
    dim3 block(kThreads);        // 16 waves (wave32)
    plrnn_step_fused<<<grid, block, kSmem, stream>>>(x, z0, AW, h, alphas, thetas, tau, out);
}
